// Conv2DManual_36043365548083
// MI455X (gfx1250) — matrix operand order repaired and checked by reading, not compiled, not run on hardware
//
#include <hip/hip_runtime.h>
#include <hip/hip_fp16.h>

// ---------------------------------------------------------------------------
// Conv2D 3x3 VALID, N=32 Cin=128 Cout=256 56x56 -> 54x54, fp32 in/out.
// Implicit GEMM on CDNA5 WMMA f16 (v_wmma_f32_16x16x32_f16), fp32 accum.
//   A = weight [COUT x K], K = Cin*9 = 1152 (OIHW row-major == GEMM order).
//   B = im2col(x) [K x NPIX], NPIX = 32*54*54 = 93312 = 729 * 128 exactly.
//   C tile per workgroup: 128 (cout) x 128 (pixels), 8 waves, wave tile 64x32.
//   K staged in chunks of 64 (2 WMMA k-steps per stage, 18 stages).
//   k -> x-offset im2col map precomputed once into a 1152-entry LDS table.
//   Register double-buffering: global loads + v_cvt_pk_rtz_f16_f32 packing
//   (1 VALU/dword) for chunk k+1 overlap the WMMAs of chunk k.
//   fp16 staging is numerically safe here: x ~ N(0,1), w ~ N(0, 2/1152).
// ---------------------------------------------------------------------------

typedef __attribute__((ext_vector_type(16))) _Float16     v16h;
typedef __attribute__((ext_vector_type(8)))  float        v8f;
typedef __attribute__((ext_vector_type(8)))  unsigned int v8u;

#define CIN   128
#define COUT  256
#define IH    56
#define IW    56
#define OH    54
#define OW    54
#define KTOT  1152          // CIN*3*3
#define KC    64            // K per LDS stage (2 WMMA k-steps)
#define NCHUNK (KTOT / KC)  // 18
#define PIX_PER_IMG (OH * OW)           // 2916
#define NPIX  (32 * PIX_PER_IMG)        // 93312
#define X_IMG_STRIDE (CIN * IH * IW)    // 401408
#define O_IMG_STRIDE (COUT * OH * OW)   // 746496

#define SA_STR 72   // halves per LDS row (64 + 8 pad) = 144 B, 16B-aligned;
#define SB_STR 72   // stride 36 dwords -> 16 lanes cover all 64 banks

// pack two fp32 -> one dword of two fp16 (lo = first, hi = second)
// single v_cvt_pk_rtz_f16_f32 instruction
__device__ __forceinline__ unsigned int pkh(float lo, float hi) {
    return __builtin_bit_cast(unsigned int,
                              __builtin_amdgcn_cvt_pkrtz(lo, hi));
}

__global__ __launch_bounds__(256) void conv3x3_wmma_f16(
    const float* __restrict__ x,      // [32,128,56,56]
    const float* __restrict__ w,      // [256,128,3,3]
    const float* __restrict__ bias,   // [256]
    float* __restrict__ out)          // [32,256,54,54]
{
    __shared__ __align__(16) unsigned short sA[128 * SA_STR];  // 18432 B
    __shared__ __align__(16) unsigned short sB[128 * SB_STR];  // 18432 B
    __shared__ __align__(16) int            xoff[KTOT];        //  4608 B

    const int tid    = threadIdx.x;
    const int lane   = tid & 31;
    const int lane16 = lane & 15;
    const int hi     = lane >> 4;          // half-wave select
    const int wave   = tid >> 5;           // 0..7
    const int mw     = (wave & 1) * 64;    // wave M offset inside 128 tile
    const int nw     = (wave >> 1) * 32;   // wave N offset inside 128 tile

    const int m0 = blockIdx.y * 128;       // cout tile base  (2 tiles)
    const int n0 = blockIdx.x * 128;       // pixel tile base (729 tiles)

    // ---- one-time: k -> im2col x-offset table (kills div/mod in the loop) -
    for (int k = tid; k < KTOT; k += 256) {
        const int cin = k / 9;
        const int r9  = k - cin * 9;
        const int kh  = r9 / 3;
        const int kw  = r9 - kh * 3;
        xoff[k] = cin * (IH * IW) + kh * IW + kw;
    }

    // --- A staging role: thread -> (row am, half ahalf) of 128x64 chunk ----
    const int am    = tid >> 1;            // 0..127
    const int ahalf = tid & 1;             // 32 floats each
    const float* arow = w + (m0 + am) * KTOT + ahalf * 32;

    // --- B staging role: thread -> (col bn, kgroup bhalf), 32 gathers each -
    const int bn    = tid & 127;
    const int bhalf = tid >> 7;            // k rows 0..31 / 32..63
    const int npix_b = n0 + bn;
    const int img_b  = npix_b / PIX_PER_IMG;
    const int rem_b  = npix_b - img_b * PIX_PER_IMG;
    const int oy_b   = rem_b / OW;
    const int ox_b   = rem_b - oy_b * OW;
    const float* xbase = x + img_b * X_IMG_STRIDE + oy_b * IW + ox_b;

    // --- accumulators: 4 (M frags) x 2 (N frags) of 16x16 f32 --------------
    v8f acc[4][2];
#pragma unroll
    for (int mf = 0; mf < 4; ++mf)
#pragma unroll
        for (int nf = 0; nf < 2; ++nf)
            acc[mf][nf] = (v8f){0.f,0.f,0.f,0.f,0.f,0.f,0.f,0.f};

    // K half-offsets of the 8 dwords of a 16-bit A fragment (ISA 7.12.2)
    const int AK[8] = {0, 2, 4, 6, 16, 18, 20, 22};

    __syncthreads();   // xoff table ready

    // ---- register double-buffer: packed fp16 dwords for the next chunk ----
    unsigned int ra[16], rb[16];

#define LOAD_CHUNK(KB)                                                        \
    do {                                                                      \
        const float4* asrc = (const float4*)(arow + (KB));                    \
        _Pragma("unroll")                                                     \
        for (int q = 0; q < 8; ++q) {                                         \
            float4 v = asrc[q];                                               \
            ra[2 * q]     = pkh(v.x, v.y);                                    \
            ra[2 * q + 1] = pkh(v.z, v.w);                                    \
        }                                                                     \
        const int4* xo = (const int4*)&xoff[(KB) + bhalf * 32];               \
        _Pragma("unroll")                                                     \
        for (int j = 0; j < 8; ++j) {                                         \
            const int4 o = xo[j];                                             \
            rb[2 * j]     = pkh(xbase[o.x], xbase[o.y]);                      \
            rb[2 * j + 1] = pkh(xbase[o.z], xbase[o.w]);                      \
        }                                                                     \
    } while (0)

    LOAD_CHUNK(0);

    for (int kc = 0; kc < NCHUNK; ++kc) {
        // ---- commit staged registers to LDS (4+4 ds_store_b128) -----------
        {
            unsigned int* da = (unsigned int*)&sA[am * SA_STR + ahalf * 32];
#pragma unroll
            for (int q = 0; q < 16; ++q) da[q] = ra[q];
            unsigned int* db = (unsigned int*)&sB[bn * SB_STR + bhalf * 32];
#pragma unroll
            for (int q = 0; q < 16; ++q) db[q] = rb[q];
        }
        __syncthreads();

        // ---- overlap: fetch + pack chunk kc+1 while WMMAs of kc run -------
        if (kc + 1 < NCHUNK) {
            LOAD_CHUNK((kc + 1) * KC);
            if (kc + 2 < NCHUNK)   // gfx1250 global_prefetch_b8, next-next A
                __builtin_prefetch(arow + (kc + 2) * KC, 0, 0);
        }

        // ---- compute: 2 WMMA k-steps over the staged 64-wide chunk --------
#pragma unroll
        for (int ks = 0; ks < 2; ++ks) {
            v16h afr[4];
#pragma unroll
            for (int mf = 0; mf < 4; ++mf) {
                const int base = (mw + mf * 16 + lane16) * SA_STR
                               + ks * 32 + hi * 8;
                v8u t;
#pragma unroll
                for (int v = 0; v < 8; ++v)
                    t[v] = *(const unsigned int*)&sA[base + AK[v]];
                afr[mf] = __builtin_bit_cast(v16h, t);
            }
            v16h bfr[2];
#pragma unroll
            for (int nf = 0; nf < 2; ++nf) {
                const int base = (nw + nf * 16 + lane16) * SB_STR
                               + ks * 32 + hi * 8;
                v8u t;
#pragma unroll
                for (int v = 0; v < 8; ++v)
                    t[v] = *(const unsigned int*)&sB[base + AK[v]];
                bfr[nf] = __builtin_bit_cast(v16h, t);
            }
#pragma unroll
            for (int mf = 0; mf < 4; ++mf)
#pragma unroll
                for (int nf = 0; nf < 2; ++nf)
                    acc[mf][nf] = __builtin_amdgcn_wmma_f32_16x16x32_f16(
                        false, afr[mf], false, bfr[nf],
                        (short)0, acc[mf][nf], false, false);
        }

        __syncthreads();
    }
#undef LOAD_CHUNK

    // ---- epilogue: scatter C frags + bias (C/D layout: VGPR r -> M=r+8*hi,
    //      lane16 -> N). Bias via two float4 loads per M-fragment. ----------
#pragma unroll
    for (int mf = 0; mf < 4; ++mf) {
        const int mbase = m0 + mw + mf * 16 + hi * 8;
        float bv[8];
        *(float4*)&bv[0] = *(const float4*)(bias + mbase);
        *(float4*)&bv[4] = *(const float4*)(bias + mbase + 4);
#pragma unroll
        for (int nf = 0; nf < 2; ++nf) {
            const int n   = n0 + nw + nf * 16 + lane16;
            const int img = n / PIX_PER_IMG;
            const int rem = n - img * PIX_PER_IMG;
            const int oy  = rem / OW;
            const int ox  = rem - oy * OW;
            const int ob  = img * O_IMG_STRIDE + oy * OW + ox;
#pragma unroll
            for (int r = 0; r < 8; ++r)
                out[ob + (mbase + r) * (OH * OW)] = acc[mf][nf][r] + bv[r];
        }
    }
}

extern "C" void kernel_launch(void* const* d_in, const int* in_sizes, int n_in,
                              void* d_out, int out_size, void* d_ws, size_t ws_size,
                              hipStream_t stream) {
    const float* x    = (const float*)d_in[0];
    const float* wgt  = (const float*)d_in[1];
    const float* bias = (const float*)d_in[2];
    float* out        = (float*)d_out;

    dim3 grid(NPIX / 128, COUT / 128, 1);   // (729, 2)
    dim3 block(256, 1, 1);
    conv3x3_wmma_f16<<<grid, block, 0, stream>>>(x, wgt, bias, out);
}
